// CustomTransformer_23029614641425
// MI455X (gfx1250) — compile-verified, matrix operand order checked by reading, not run on hardware
//
#include <hip/hip_runtime.h>
#include <math.h>

// ---------------- problem constants (match reference) ----------------
#define E   512
#define Hh  8
#define Dd  64
#define FF  2048
#define Bb  8
#define Ss  1024
#define Tt  1024
#define M_ROWS (Bb * Ss)   // 8192 token rows (S == T)
#define LDQKV (3 * E)      // 1536, compile-time stride of the QKV buffer
#define NCH (Tt / 32)      // 32 key chunks
#define VT_STRIDE 40       // padded bf16 row stride for V^T LDS tile (16B aligned)
#define VT_SZ (64 * VT_STRIDE)

typedef __attribute__((ext_vector_type(16))) __bf16 bf16x16;
typedef __attribute__((ext_vector_type(8)))  __bf16 bf16x8;
typedef __attribute__((ext_vector_type(4)))  __bf16 bf16x4;
typedef __attribute__((ext_vector_type(2)))  __bf16 bf16x2;
typedef __attribute__((ext_vector_type(8)))  float  f32x8;

// ---------------- WMMA helper ----------------
__device__ __forceinline__ f32x8 wmma_bf16(bf16x16 a, bf16x16 b, f32x8 c) {
  return __builtin_amdgcn_wmma_f32_16x16x32_bf16(false, a, false, b, (short)0, c,
                                                 false, false);
}

// lane-xor-16 swap via v_permlanex16_b32 (pure VALU, no LDS round-trip)
__device__ __forceinline__ float xswap16(float x) {
  const int i = __builtin_bit_cast(int, x);
  const int r = __builtin_amdgcn_permlanex16(i, i, 0x76543210, 0xfedcba98,
                                             false, false);
  return __builtin_bit_cast(float, r);
}

// One lane's 16-element run of a 16x32 bf16 A/B fragment (two 16B loads).
__device__ __forceinline__ bf16x16 frag_ld_h(const __bf16* __restrict__ p) {
  bf16x8 lo = *reinterpret_cast<const bf16x8*>(p);
  bf16x8 hi = *reinterpret_cast<const bf16x8*>(p + 16);
  return __builtin_shufflevector(lo, hi, 0, 1, 2, 3, 4, 5, 6, 7, 8, 9, 10, 11,
                                 12, 13, 14, 15);
}

// ---------------- GEMM: C[M,N] = act(A[M,K] @ W[N,K]^T + bias[N]) ----------------
// bf16 in; 256 threads = 8 waves; block tile 128x128; wave tile 64x32.
// Software-pipelined: next k-step fragments prefetched during current WMMAs.
template <int RELU, int OUTBF>
__global__ __launch_bounds__(256) void gemm_h(
    const __bf16* __restrict__ A, const __bf16* __restrict__ W,
    const float* __restrict__ bias, void* __restrict__ Cv,
    int M, int N, int K, int ldc) {
  const int lane = threadIdx.x & 31;
  const int wave = threadIdx.x >> 5;
  const int m0 = blockIdx.y * 128 + (wave & 1) * 64;
  const int n0 = blockIdx.x * 128 + (wave >> 1) * 32;
  const int r16 = lane & 15;
  const int kb8 = (lane >> 4) << 3;

  f32x8 acc[4][2] = {};
  const __bf16* Ar[4];
  const __bf16* Wr[2];
#pragma unroll
  for (int mt = 0; mt < 4; ++mt)
    Ar[mt] = A + (size_t)(m0 + mt * 16 + r16) * K + kb8;
#pragma unroll
  for (int nt = 0; nt < 2; ++nt)
    Wr[nt] = W + (size_t)(n0 + nt * 16 + r16) * K + kb8;

  bf16x16 cb0 = frag_ld_h(Wr[0]);
  bf16x16 cb1 = frag_ld_h(Wr[1]);
  bf16x16 ca[4];
#pragma unroll
  for (int mt = 0; mt < 4; ++mt) ca[mt] = frag_ld_h(Ar[mt]);

  for (int k0 = 32; k0 < K; k0 += 32) {
    bf16x16 nb0 = frag_ld_h(Wr[0] + k0);
    bf16x16 nb1 = frag_ld_h(Wr[1] + k0);
    bf16x16 na[4];
#pragma unroll
    for (int mt = 0; mt < 4; ++mt) na[mt] = frag_ld_h(Ar[mt] + k0);
#pragma unroll
    for (int mt = 0; mt < 4; ++mt) {
      acc[mt][0] = wmma_bf16(ca[mt], cb0, acc[mt][0]);
      acc[mt][1] = wmma_bf16(ca[mt], cb1, acc[mt][1]);
    }
    cb0 = nb0;
    cb1 = nb1;
#pragma unroll
    for (int mt = 0; mt < 4; ++mt) ca[mt] = na[mt];
  }
#pragma unroll
  for (int mt = 0; mt < 4; ++mt) {
    acc[mt][0] = wmma_bf16(ca[mt], cb0, acc[mt][0]);
    acc[mt][1] = wmma_bf16(ca[mt], cb1, acc[mt][1]);
  }

  float bval[2];
  bval[0] = bias[n0 + r16];
  bval[1] = bias[n0 + 16 + r16];

#pragma unroll
  for (int mt = 0; mt < 4; ++mt)
#pragma unroll
    for (int nt = 0; nt < 2; ++nt)
#pragma unroll
      for (int r = 0; r < 8; ++r) {
        const int m = m0 + mt * 16 + kb8 + r;  // C: M = r + 8*(lane>>4)
        const int n = n0 + nt * 16 + r16;      //    N = lane&15
        float v = acc[mt][nt][r] + bval[nt];
        if (RELU) v = fmaxf(v, 0.0f);
        if (OUTBF)
          ((__bf16*)Cv)[(size_t)m * ldc + n] = (__bf16)v;
        else
          ((float*)Cv)[(size_t)m * ldc + n] = v;
      }
}

// ---------------- Gaussian attention (flash-style, one pass, bf16 QKV) ----------
// S^T = K@Q^T -> queries on lanes -> lane-local softmax; P^T registers are the
// B-fragment for O^T = V^T @ P^T. Double-buffered LDS V^T + prefetched K frags.
// Gaussian bias folded into the exp argument via a float distance recurrence.
__global__ __launch_bounds__(256) void gauss_attn(
    const __bf16* __restrict__ Qb, const __bf16* __restrict__ Kb,
    const __bf16* __restrict__ Vb, __bf16* __restrict__ O, float inv2sig) {
  __shared__ __bf16 vt[2 * VT_SZ];

  const int lane = threadIdx.x & 31;
  const int wave = threadIdx.x >> 5;
  const int b = blockIdx.y / Hh;
  const int h = blockIdx.y % Hh;
  const int q0 = blockIdx.x * 128 + wave * 16;
  const int r16 = lane & 15;
  const int kb8 = (lane >> 4) << 3;

  // Q fragments (B operand), loaded once.
  const __bf16* Qrow = Qb + (size_t)(b * Tt + q0 + r16) * LDQKV + h * Dd + kb8;
  const bf16x16 qf0 = frag_ld_h(Qrow);
  const bf16x16 qf1 = frag_ld_h(Qrow + 32);

  // V staging: thread owns key pair {2kp,2kp+1} x 4 d-values -> paired b32 stores
  const int kp2 = (threadIdx.x & 15) * 2;   // 0..30
  const int d4 = (threadIdx.x >> 4) * 4;    // 0..60
  const __bf16* Vp0 = Vb + (size_t)(b * Tt + kp2) * LDQKV + h * Dd + d4;
  const int soff = d4 * VT_STRIDE + kp2;    // LDS store base (element units)
  const int vread = r16 * VT_STRIDE + kb8;  // LDS read base (per lane)

  // K fragment base (advanced additively; tile/k offsets become immediates)
  const __bf16* Kcur = Kb + (size_t)(b * Tt + r16) * LDQKV + h * Dd + kb8;

  const int qidx = q0 + r16;
  float mrun = -3.0e38f, l_e = 0.0f, l_eg = 0.0f;
  f32x8 acc[4] = {};  // O^T: 4 d-tiles of 16; lane-local query column

  // per-element signed distances q-key (float recurrence, -32 per chunk)
  float dds[2][8];
#pragma unroll
  for (int t = 0; t < 2; ++t)
#pragma unroll
    for (int r = 0; r < 8; ++r)
      dds[t][r] = (float)(qidx - (t * 16 + kb8 + r));

  // ---- prologue: stage chunk 0 (LDS buf 0) and load its K fragments ----
  {
    bf16x4 va = *reinterpret_cast<const bf16x4*>(Vp0);
    bf16x4 vb2 = *reinterpret_cast<const bf16x4*>(Vp0 + LDQKV);
#pragma unroll
    for (int c = 0; c < 4; ++c) {
      bf16x2 pr;
      pr[0] = va[c];
      pr[1] = vb2[c];
      *reinterpret_cast<bf16x2*>(&vt[soff + c * VT_STRIDE]) = pr;
    }
  }
  bf16x16 kf[4];
  kf[0] = frag_ld_h(Kcur);
  kf[1] = frag_ld_h(Kcur + 32);
  kf[2] = frag_ld_h(Kcur + 16 * LDQKV);
  kf[3] = frag_ld_h(Kcur + 16 * LDQKV + 32);
  __syncthreads();

#pragma unroll 2
  for (int i = 0; i < NCH; ++i) {
    const bool more = (i + 1 < NCH);

    // ---- prefetch next chunk: K fragments (regs) + V^T (other LDS buffer) ----
    bf16x16 nk[4];
    if (more) {
      Kcur += 32 * LDQKV;
      nk[0] = frag_ld_h(Kcur);
      nk[1] = frag_ld_h(Kcur + 32);
      nk[2] = frag_ld_h(Kcur + 16 * LDQKV);
      nk[3] = frag_ld_h(Kcur + 16 * LDQKV + 32);
      Vp0 += 32 * LDQKV;
      bf16x4 va = *reinterpret_cast<const bf16x4*>(Vp0);
      bf16x4 vb2 = *reinterpret_cast<const bf16x4*>(Vp0 + LDQKV);
      __bf16* dst = &vt[((i + 1) & 1) * VT_SZ + soff];
#pragma unroll
      for (int c = 0; c < 4; ++c) {
        bf16x2 pr;
        pr[0] = va[c];
        pr[1] = vb2[c];
        *reinterpret_cast<bf16x2*>(dst + c * VT_STRIDE) = pr;
      }
    }

    // ---- scores S^T (raw; 1/sqrt(D) folded into exp arg) ----
    float s[2][8];
#pragma unroll
    for (int t = 0; t < 2; ++t) {
      f32x8 sc = {};
      sc = wmma_bf16(kf[2 * t], qf0, sc);
      sc = wmma_bf16(kf[2 * t + 1], qf1, sc);
#pragma unroll
      for (int r = 0; r < 8; ++r) s[t][r] = sc[r];
    }

    // ---- online softmax with gaussian bias fused into exp ----
    float cmax = -3.0e38f;
#pragma unroll
    for (int t = 0; t < 2; ++t)
#pragma unroll
      for (int r = 0; r < 8; ++r) cmax = fmaxf(cmax, s[t][r]);
    cmax = fmaxf(cmax, xswap16(cmax));
    const float mnew = fmaxf(mrun, cmax * 0.125f);
    const float corr = __expf(mrun - mnew);

    float le_c = 0.0f, leg_c = 0.0f;
    float p[2][8];
#pragma unroll
    for (int t = 0; t < 2; ++t)
#pragma unroll
      for (int r = 0; r < 8; ++r) {
        const float earg = fmaf(s[t][r], 0.125f, -mnew);
        const float dd = dds[t][r];
        const float parg = fmaf(dd * dd, -inv2sig, earg);
        const float e = __expf(earg);   // softmax numerator
        const float pp = __expf(parg);  // numerator * gaussian
        le_c += e;
        leg_c += pp;
        p[t][r] = pp;
        dds[t][r] = dd - 32.0f;
      }
    le_c += xswap16(le_c);
    leg_c += xswap16(leg_c);
    l_e = l_e * corr + le_c;
    l_eg = l_eg * corr + leg_c;
    mrun = mnew;

    bf16x16 pf;
#pragma unroll
    for (int j = 0; j < 8; ++j) {
      pf[j] = (__bf16)p[0][j];
      pf[8 + j] = (__bf16)p[1][j];
    }

    // ---- O^T += V^T(chunk) @ P^T with running rescale ----
    const __bf16* vbuf = &vt[(i & 1) * VT_SZ + vread];
#pragma unroll
    for (int tile = 0; tile < 4; ++tile) {
      bf16x8 vlo = *reinterpret_cast<const bf16x8*>(vbuf + tile * 16 * VT_STRIDE);
      bf16x8 vhi =
          *reinterpret_cast<const bf16x8*>(vbuf + tile * 16 * VT_STRIDE + 16);
      bf16x16 vf = __builtin_shufflevector(vlo, vhi, 0, 1, 2, 3, 4, 5, 6, 7, 8,
                                           9, 10, 11, 12, 13, 14, 15);
      f32x8 a = acc[tile];
#pragma unroll
      for (int r = 0; r < 8; ++r) a[r] *= corr;
      acc[tile] = wmma_bf16(vf, pf, a);
    }

    __syncthreads();
    if (more) {
#pragma unroll
      for (int t = 0; t < 4; ++t) kf[t] = nk[t];
    }
  }

  const float rden = 1.0f / (l_eg + 1e-5f * l_e);
  __bf16* Orow = O + (size_t)(b * Tt + qidx) * E + h * Dd;
#pragma unroll
  for (int tile = 0; tile < 4; ++tile) {
    bf16x8 ov;
#pragma unroll
    for (int r = 0; r < 8; ++r) ov[r] = (__bf16)(acc[tile][r] * rden);
    *reinterpret_cast<bf16x8*>(Orow + tile * 16 + kb8) = ov;  // one 16B store
  }
}

// ---------------- residual + LayerNorm: out = LN(x + y), dual fp32/bf16 ----------
__global__ __launch_bounds__(256) void add_ln(
    const float* __restrict__ X, const float* __restrict__ Y,
    const float* __restrict__ g, const float* __restrict__ bta,
    float* __restrict__ OutF, __bf16* __restrict__ OutH) {
  __shared__ float red[16];
  const int row = blockIdx.x;
  const int t = threadIdx.x;
  const size_t base = (size_t)row * E;
  const float v0 = X[base + t] + Y[base + t];
  const float v1 = X[base + t + 256] + Y[base + t + 256];
  float s = v0 + v1, s2 = v0 * v0 + v1 * v1;
#pragma unroll
  for (int o = 16; o >= 1; o >>= 1) {
    s += __shfl_xor(s, o, 32);
    s2 += __shfl_xor(s2, o, 32);
  }
  const int wv = t >> 5;
  if ((t & 31) == 0) {
    red[wv] = s;
    red[8 + wv] = s2;
  }
  __syncthreads();
  float ts = 0.0f, ts2 = 0.0f;
#pragma unroll
  for (int i = 0; i < 8; ++i) {
    ts += red[i];
    ts2 += red[8 + i];
  }
  const float mean = ts * (1.0f / (float)E);
  const float var = ts2 * (1.0f / (float)E) - mean * mean;
  const float rstd = rsqrtf(var + 1e-5f);
  const float o0 = (v0 - mean) * rstd * g[t] + bta[t];
  const float o1 = (v1 - mean) * rstd * g[t + 256] + bta[t + 256];
  OutF[base + t] = o0;
  OutF[base + t + 256] = o1;
  OutH[base + t] = (__bf16)o0;
  OutH[base + t + 256] = (__bf16)o1;
}

// ---------------- positional encoding add (batch-indexed, bug-faithful) ----------
__global__ void add_pos(const float* __restrict__ src, float* __restrict__ out,
                        __bf16* __restrict__ outh, int total) {
  const int i = blockIdx.x * blockDim.x + threadIdx.x;
  if (i >= total) return;
  const int e = i & (E - 1);
  const int b = (i / E) / Ss;  // pe indexed by batch in the reference
  const int half = e >> 1;
  const float div = __expf((float)(2 * half) * (-logf(10000.0f) / (float)E));
  const float ang = (float)b * div;
  const float pe = (e & 1) ? __cosf(ang) : __sinf(ang);
  const float v = src[i] + pe;
  out[i] = v;
  outh[i] = (__bf16)v;
}

// ---------------- fp32 -> bf16 convert (vectorized x4) ----------------
__global__ void f2h4(const float* __restrict__ in, __bf16* __restrict__ out,
                     int n4) {
  const int i = blockIdx.x * blockDim.x + threadIdx.x;
  if (i >= n4) return;
  const float4 v = reinterpret_cast<const float4*>(in)[i];
  bf16x4 o;
  o[0] = (__bf16)v.x;
  o[1] = (__bf16)v.y;
  o[2] = (__bf16)v.z;
  o[3] = (__bf16)v.w;
  reinterpret_cast<bf16x4*>(out)[i] = o;
}

// ---------------- host orchestration ----------------
extern "C" void kernel_launch(void* const* d_in, const int* in_sizes, int n_in,
                              void* d_out, int out_size, void* d_ws,
                              size_t ws_size, hipStream_t stream) {
  (void)in_sizes; (void)n_in; (void)out_size; (void)ws_size;
  const float* src = (const float*)d_in[0];
  const float* tgt = (const float*)d_in[1];

  const size_t NE = (size_t)M_ROWS * E;
  float* buf_x = (float*)d_ws;           // fp32 residual state (encoder / memory)
  float* buf_u = buf_x + NE;             // fp32 branch output
  float* buf_y = buf_u + NE;             // fp32 decoder state
  __bf16* hbase = (__bf16*)(buf_y + NE);
  __bf16* buf_xh = hbase;                // bf16 enc state
  __bf16* buf_yh = buf_xh + NE;          // bf16 dec state (also tgt copy)
  __bf16* buf_memh = buf_yh + NE;        // bf16 encoder memory
  __bf16* buf_th = buf_memh + NE;        // bf16 attention output
  __bf16* buf_qkvh = buf_th + NE;        // bf16 [8192,1536]
  __bf16* buf_ffh = buf_qkvh + NE * 3;   // bf16 [8192,2048]
  __bf16* wh = buf_ffh + (size_t)M_ROWS * FF;  // bf16 weight arena

  auto conv = [&](const float* in, __bf16* out, size_t n) {
    const int n4 = (int)(n / 4);
    f2h4<<<dim3((n4 + 255) / 256), dim3(256), 0, stream>>>(in, out, n4);
  };
  auto gemm = [&](const __bf16* A, const __bf16* W, const float* bias, void* C,
                  int M, int N, int K, int ldc, bool relu, bool outbf) {
    dim3 g(N / 128, M / 128), blk(256);
    if (outbf) {
      if (relu) gemm_h<1, 1><<<g, blk, 0, stream>>>(A, W, bias, C, M, N, K, ldc);
      else      gemm_h<0, 1><<<g, blk, 0, stream>>>(A, W, bias, C, M, N, K, ldc);
    } else {
      if (relu) gemm_h<1, 0><<<g, blk, 0, stream>>>(A, W, bias, C, M, N, K, ldc);
      else      gemm_h<0, 0><<<g, blk, 0, stream>>>(A, W, bias, C, M, N, K, ldc);
    }
  };
  const float inv2sig = 1.0f / (2.0f * 256.0f * 256.0f);  // sigma = Tq/4 = 256
  auto attn = [&](const __bf16* qkv, __bf16* O) {
    dim3 g(Tt / 128, Bb * Hh), blk(256);
    gauss_attn<<<g, blk, 0, stream>>>(qkv, qkv + E, qkv + 2 * E, O, inv2sig);
  };
  auto lnadd = [&](const float* X, const float* Y, const float* g_,
                   const float* b_, float* OF, __bf16* OH) {
    add_ln<<<dim3(M_ROWS), dim3(256), 0, stream>>>(X, Y, g_, b_, OF, OH);
  };

  // ---- convert all weights to bf16 once ----
  __bf16* ewh[3][4];
  __bf16* dwh[3][6];
  {
    __bf16* cur = wh;
    for (int L = 0; L < 3; ++L) {
      const int p = 2 + L * 12;
      const size_t sz[4] = {(size_t)3 * E * E, (size_t)E * E, (size_t)FF * E,
                            (size_t)E * FF};
      const int idx[4] = {0, 2, 8, 10};  // Wi, Wo, W1, W2
      for (int k = 0; k < 4; ++k) {
        ewh[L][k] = cur;
        conv((const float*)d_in[p + idx[k]], cur, sz[k]);
        cur += sz[k];
      }
    }
    for (int L = 0; L < 3; ++L) {
      const int p = 2 + 36 + L * 18;
      const size_t sz[6] = {(size_t)3 * E * E, (size_t)E * E, (size_t)3 * E * E,
                            (size_t)E * E,     (size_t)FF * E, (size_t)E * FF};
      const int idx[6] = {0, 2, 4, 6, 14, 16};  // sWi, sWo, cWi, cWo, W1, W2
      for (int k = 0; k < 6; ++k) {
        dwh[L][k] = cur;
        conv((const float*)d_in[p + idx[k]], cur, sz[k]);
        cur += sz[k];
      }
    }
  }

  add_pos<<<dim3((M_ROWS * E) / 256), dim3(256), 0, stream>>>(
      src, buf_x, buf_xh, M_ROWS * E);

  // -------- encoder --------
  for (int L = 0; L < 3; ++L) {
    const int p = 2 + L * 12;
    const float* bi = (const float*)d_in[p + 1];
    const float* bo = (const float*)d_in[p + 3];
    const float* g1 = (const float*)d_in[p + 4];
    const float* b1 = (const float*)d_in[p + 5];
    const float* g2 = (const float*)d_in[p + 6];
    const float* b2 = (const float*)d_in[p + 7];
    const float* fb1 = (const float*)d_in[p + 9];
    const float* fb2 = (const float*)d_in[p + 11];

    gemm(buf_xh, ewh[L][0], bi, buf_qkvh, M_ROWS, 3 * E, E, 3 * E, false, true);
    attn(buf_qkvh, buf_th);
    gemm(buf_th, ewh[L][1], bo, buf_u, M_ROWS, E, E, E, false, false);
    lnadd(buf_x, buf_u, g1, b1, buf_x, buf_xh);
    gemm(buf_xh, ewh[L][2], fb1, buf_ffh, M_ROWS, FF, E, FF, true, true);
    gemm(buf_ffh, ewh[L][3], fb2, buf_u, M_ROWS, E, FF, E, false, false);
    lnadd(buf_x, buf_u, g2, b2, buf_x, buf_xh);
  }

  // encoder memory (bf16) for cross-attention K/V projections
  conv(buf_x, buf_memh, NE);
  // decoder initial state: tgt (fp32 stays in d_in, bf16 copy for GEMMs)
  conv(tgt, buf_yh, NE);

  // -------- decoder --------
  const float* ycur = tgt;
  for (int L = 0; L < 3; ++L) {
    const int p = 2 + 36 + L * 18;
    const float* sbi = (const float*)d_in[p + 1];
    const float* sbo = (const float*)d_in[p + 3];
    const float* cbi = (const float*)d_in[p + 5];
    const float* cbo = (const float*)d_in[p + 7];
    const float* g1 = (const float*)d_in[p + 8];
    const float* b1 = (const float*)d_in[p + 9];
    const float* g2 = (const float*)d_in[p + 10];
    const float* b2 = (const float*)d_in[p + 11];
    const float* g3 = (const float*)d_in[p + 12];
    const float* b3 = (const float*)d_in[p + 13];
    const float* fb1 = (const float*)d_in[p + 15];
    const float* fb2 = (const float*)d_in[p + 17];

    // self-attention
    gemm(buf_yh, dwh[L][0], sbi, buf_qkvh, M_ROWS, 3 * E, E, 3 * E, false, true);
    attn(buf_qkvh, buf_th);
    gemm(buf_th, dwh[L][1], sbo, buf_u, M_ROWS, E, E, E, false, false);
    lnadd(ycur, buf_u, g1, b1, buf_y, buf_yh);

    // cross-attention: Q from y, K/V from encoder memory
    gemm(buf_yh, dwh[L][2], cbi, buf_qkvh, M_ROWS, E, E, 3 * E, false, true);
    gemm(buf_memh, dwh[L][2] + (size_t)E * E, cbi + E, buf_qkvh + E, M_ROWS,
         2 * E, E, 3 * E, false, true);
    attn(buf_qkvh, buf_th);
    gemm(buf_th, dwh[L][3], cbo, buf_u, M_ROWS, E, E, E, false, false);
    lnadd(buf_y, buf_u, g2, b2, buf_y, buf_yh);

    // FFN
    gemm(buf_yh, dwh[L][4], fb1, buf_ffh, M_ROWS, FF, E, FF, true, true);
    gemm(buf_ffh, dwh[L][5], fb2, buf_u, M_ROWS, E, FF, E, false, false);
    float* outp = (L == 2) ? (float*)d_out : buf_y;
    lnadd(buf_y, buf_u, g3, b3, outp, buf_yh);
    ycur = buf_y;
  }
}
